// LikelihoodRatioEstimator_61048665145506
// MI455X (gfx1250) — matrix operand order repaired and checked by reading, not compiled, not run on hardware
//
#include <hip/hip_runtime.h>

typedef __attribute__((ext_vector_type(16))) __bf16 v16bf;
typedef __attribute__((ext_vector_type(8)))  float  v8f;

union V16 { uint4 q[2]; v16bf v; };

#define D_DIM     128
#define BLK       128      // output tile edge
#define ROWPITCH  136      // ushorts per LDS row (128 data + 8 pad -> conflict-free b128)
#define LN2       0.69314718055994530942f

// ---- CDNA5 async global->LDS path (guarded; falls back to load+ds_store) ----
#if defined(__has_builtin)
#if __has_builtin(__builtin_amdgcn_global_load_async_to_lds_b128)
#define LRE_ASYNC_LDS 1
#endif
#endif

#ifdef LRE_ASYNC_LDS
// builtin signature (from hipcc diagnostic): (v4i addrspace(1)*, v4i addrspace(3)*, imm, imm)
typedef int lre_i4 __attribute__((__vector_size__(16)));
typedef __attribute__((address_space(1))) lre_i4 lre_gi4;   // global-pointee
typedef __attribute__((address_space(3))) lre_i4 lre_li4;   // lds-pointee

static __device__ __forceinline__ void lre_async_b128(const void* gptr, void* lptr) {
    __builtin_amdgcn_global_load_async_to_lds_b128((lre_gi4*)gptr, (lre_li4*)lptr, 0, 0);
}
static __device__ __forceinline__ void lre_wait_async0() {
#if __has_builtin(__builtin_amdgcn_s_wait_asynccnt)
    __builtin_amdgcn_s_wait_asynccnt(0);
#else
    asm volatile("s_wait_asynccnt 0" ::: "memory");
#endif
}
#endif

// ---------- helpers: f32 <-> bf16 (RNE) ----------
static __device__ __forceinline__ unsigned short f2bf(float f) {
    unsigned int u = __float_as_uint(f);
    unsigned int r = u + 0x7FFFu + ((u >> 16) & 1u);
    return (unsigned short)(r >> 16);
}
static __device__ __forceinline__ float bf2f(unsigned short h) {
    return __uint_as_float(((unsigned int)h) << 16);
}

// ---------- kernel 1: split f32 -> bf16 hi/lo ----------
__global__ void lre_split_kernel(const float* __restrict__ zx,
                                 const float* __restrict__ zy,
                                 unsigned short* __restrict__ Xhi,
                                 unsigned short* __restrict__ Xlo,
                                 unsigned short* __restrict__ Yhi,
                                 unsigned short* __restrict__ Ylo,
                                 int nd) {
    int i = blockIdx.x * blockDim.x + threadIdx.x;
    if (i >= 2 * nd) return;
    bool isX = (i < nd);
    int j = isX ? i : (i - nd);
    float v = isX ? zx[j] : zy[j];
    unsigned short hi = f2bf(v);
    unsigned short lo = f2bf(v - bf2f(hi));
    if (isX) { Xhi[j] = hi; Xlo[j] = lo; }
    else     { Yhi[j] = hi; Ylo[j] = lo; }
}

// ---------- kernel 2: row squared norms ----------
__global__ void lre_norms_kernel(const float* __restrict__ zx,
                                 const float* __restrict__ zy,
                                 float* __restrict__ xx,
                                 float* __restrict__ yy,
                                 int n) {
    int i = blockIdx.x * blockDim.x + threadIdx.x;
    if (i >= 2 * n) return;
    bool isX = (i < n);
    int r = isX ? i : (i - n);
    const float* p = (isX ? zx : zy) + (size_t)r * D_DIM;
    float s = 0.f;
#pragma unroll 8
    for (int k = 0; k < D_DIM; ++k) s += p[k] * p[k];
    if (isX) xx[r] = s; else yy[r] = s;
}

// ---------- kernel 3: tiled WMMA GEMM + fused epilogue/reduction ----------
__global__ __launch_bounds__(256)
void lre_gemm_kernel(const unsigned short* __restrict__ Xhi,
                     const unsigned short* __restrict__ Xlo,
                     const unsigned short* __restrict__ Yhi,
                     const unsigned short* __restrict__ Ylo,
                     const float* __restrict__ xx,
                     const float* __restrict__ yy,
                     float* __restrict__ partials,
                     int n) {
    extern __shared__ unsigned short lds_raw[];
    unsigned short* Yh_s = lds_raw;                    // 128 x ROWPITCH bf16
    unsigned short* Yl_s = lds_raw + BLK * ROWPITCH;   // 128 x ROWPITCH bf16

    const int tid   = threadIdx.x;
    const int w     = tid >> 5;        // wave id (8 waves, wave32)
    const int lane  = tid & 31;
    const int lhalf = lane >> 4;       // 0 | 1 (lane half selects K sub-chunk)
    const int lmod  = lane & 15;

    const int rowBase = blockIdx.y * BLK + w * 16;  // this wave's 16 X-rows
    const int colBase = blockIdx.x * BLK;           // this block's 128 Y-cols
    const bool diagBlock = (blockIdx.x == blockIdx.y);

    const unsigned short* Xh = Xhi + (size_t)(rowBase + lmod) * D_DIM;
    const unsigned short* Xl = Xlo + (size_t)(rowBase + lmod) * D_DIM;
    __builtin_prefetch(Xh, 0, 1);   // global_prefetch_b8 of this wave's A rows
    __builtin_prefetch(Xl, 0, 1);

    // ---- stage Y tile (hi+lo) into padded LDS: 2048 b128 chunks each ----
#ifdef LRE_ASYNC_LDS
    for (int c = tid; c < BLK * 16; c += 256) {
        int row = c >> 4;          // Y row within tile
        int cc  = c & 15;          // 16B chunk within row
        lre_async_b128(Yhi + (size_t)(colBase + row) * D_DIM + cc * 8,
                       Yh_s + row * ROWPITCH + cc * 8);
        lre_async_b128(Ylo + (size_t)(colBase + row) * D_DIM + cc * 8,
                       Yl_s + row * ROWPITCH + cc * 8);
    }
    lre_wait_async0();
    __syncthreads();
#else
    for (int c = tid; c < BLK * 16; c += 256) {
        int row = c >> 4;
        int cc  = c & 15;
        const uint4* sh = reinterpret_cast<const uint4*>(Yhi + (size_t)(colBase + row) * D_DIM) + cc;
        const uint4* sl = reinterpret_cast<const uint4*>(Ylo + (size_t)(colBase + row) * D_DIM) + cc;
        *reinterpret_cast<uint4*>(Yh_s + row * ROWPITCH + cc * 8) = *sh;
        *reinterpret_cast<uint4*>(Yl_s + row * ROWPITCH + cc * 8) = *sl;
    }
    __syncthreads();
#endif

    // ---- main loop: wave computes a 16x128 strip via 8 accumulators ----
    v8f acc[8] = {};

#pragma unroll
    for (int kk = 0; kk < D_DIM; kk += 32) {
        // A operand (16x32 bf16): lanes 0-15 row=lmod K=[kk..kk+15];
        // lanes 16-31 K=[kk+16..kk+31]
        const int c0 = kk + 16 * lhalf;
        const int c1 = kk + 16 * lhalf + 8;
        V16 ah, al;
        ah.q[0] = *reinterpret_cast<const uint4*>(Xh + c0);
        ah.q[1] = *reinterpret_cast<const uint4*>(Xh + c1);
        al.q[0] = *reinterpret_cast<const uint4*>(Xl + c0);
        al.q[1] = *reinterpret_cast<const uint4*>(Xl + c1);

#pragma unroll
        for (int t = 0; t < 8; ++t) {
            // B operand (32x16 bf16): lane holds 32 contiguous bytes of Y row
            // (col = lmod + 16t), K base = kk + 16*lhalf
            const unsigned short* bh = Yh_s + (lmod + 16 * t) * ROWPITCH + kk + 16 * lhalf;
            const unsigned short* bl = Yl_s + (lmod + 16 * t) * ROWPITCH + kk + 16 * lhalf;
            V16 vbh, vbl;
            vbh.q[0] = *reinterpret_cast<const uint4*>(bh);
            vbh.q[1] = *reinterpret_cast<const uint4*>(bh + 8);
            vbl.q[0] = *reinterpret_cast<const uint4*>(bl);
            vbl.q[1] = *reinterpret_cast<const uint4*>(bl + 8);
            // split-bf16: dot ~= hi*hi + hi*lo + lo*hi  (f32 accumulate)
            acc[t] = __builtin_amdgcn_wmma_f32_16x16x32_bf16(false, ah.v, false, vbh.v,
                                                             (short)0, acc[t], false, false);
            acc[t] = __builtin_amdgcn_wmma_f32_16x16x32_bf16(false, ah.v, false, vbl.v,
                                                             (short)0, acc[t], false, false);
            acc[t] = __builtin_amdgcn_wmma_f32_16x16x32_bf16(false, al.v, false, vbh.v,
                                                             (short)0, acc[t], false, false);
        }
    }

    // ---- epilogue: logit = -log1p(sq), exp(logit) = 1/(1+sq) ----
    // C/D layout: lane l, vgpr r -> M = r + 8*(l>=16), Ncol = l&15
    float xr[8];
#pragma unroll
    for (int r = 0; r < 8; ++r) xr[r] = xx[rowBase + r + 8 * lhalf];

    float s_log = 0.f, s_exp = 0.f, s_dl = 0.f, s_de = 0.f;
#pragma unroll
    for (int t = 0; t < 8; ++t) {
        const int gcol = colBase + t * 16 + lmod;
        const float yv = yy[gcol];
#pragma unroll
        for (int r = 0; r < 8; ++r) {
            float dot = acc[t][r];
            float sq  = xr[r] + yv - 2.0f * dot;
            sq        = fmaxf(sq, 0.0f);
            float wv  = 1.0f + sq;
            float lg  = -LN2 * __builtin_amdgcn_logf(wv);   // -ln(1+sq)
            float ex  = __builtin_amdgcn_rcpf(wv);          // exp(logit)
            s_log += lg;
            s_exp += ex;
            // diagonal lives only in blocks bx==by, and there only in subtile t==w
            if (diagBlock && (t == w)) {
                if (r + 8 * lhalf == lmod) { s_dl += lg; s_de += ex; }
            }
        }
    }

    // ---- block reduction (wave32 shuffles, fixed order -> deterministic) ----
    float part[4] = { s_log, s_exp, s_dl, s_de };
#pragma unroll
    for (int k = 0; k < 4; ++k)
        for (int o = 16; o > 0; o >>= 1)
            part[k] += __shfl_down(part[k], o);

    __shared__ float red[8][4];
    if (lane == 0) {
#pragma unroll
        for (int k = 0; k < 4; ++k) red[w][k] = part[k];
    }
    __syncthreads();
    if (tid == 0) {
        float tot[4] = { 0.f, 0.f, 0.f, 0.f };
        for (int ww = 0; ww < 8; ++ww)
#pragma unroll
            for (int k = 0; k < 4; ++k) tot[k] += red[ww][k];
        const int bid = blockIdx.y * gridDim.x + blockIdx.x;
#pragma unroll
        for (int k = 0; k < 4; ++k) partials[bid * 4 + k] = tot[k];
    }
}

// ---------- kernel 4: deterministic final reduction + outputs ----------
__global__ void lre_finalize_kernel(const float* __restrict__ partials,
                                    int nPart, float* __restrict__ out, int n) {
    const int tid  = threadIdx.x;
    const int w    = tid >> 5;
    const int lane = tid & 31;
    double s[4] = { 0.0, 0.0, 0.0, 0.0 };
    for (int i = tid; i < nPart; i += 256)
#pragma unroll
        for (int k = 0; k < 4; ++k) s[k] += (double)partials[i * 4 + k];
#pragma unroll
    for (int k = 0; k < 4; ++k)
        for (int o = 16; o > 0; o >>= 1)
            s[k] += __shfl_down(s[k], o);

    __shared__ double red[8][4];
    if (lane == 0) {
#pragma unroll
        for (int k = 0; k < 4; ++k) red[w][k] = s[k];
    }
    __syncthreads();
    if (tid == 0) {
        double Sl = 0, Se = 0, Dl = 0, De = 0;
        for (int ww = 0; ww < 8; ++ww) {
            Sl += red[ww][0]; Se += red[ww][1]; Dl += red[ww][2]; De += red[ww][3];
        }
        double Nn       = (double)n;
        double cnt      = Nn * (Nn - 1.0);
        double mean_pos = Dl / Nn;
        double mean_neg = (Sl - Dl) / cnt;
        double se_neg   = Se - De;
        double lb       = log(se_neg) - log(cnt);           // logsumexp - log(count)
        double repuls   = (se_neg * exp(-lb)) / cnt;        // == 1 up to fp
        double loss     = (lb - mean_pos) + repuls;         // attraction + repulsion
        out[0] = (float)mean_pos;
        out[1] = (float)mean_neg;
        out[2] = (float)lb;
        out[3] = (float)loss;
    }
}

extern "C" void kernel_launch(void* const* d_in, const int* in_sizes, int n_in,
                              void* d_out, int out_size, void* d_ws, size_t ws_size,
                              hipStream_t stream) {
    (void)n_in; (void)out_size; (void)ws_size;
    const float* zx = (const float*)d_in[0];
    const float* zy = (const float*)d_in[1];
    float* out = (float*)d_out;

    const int N  = in_sizes[0] / D_DIM;   // 8192
    const int nd = N * D_DIM;

    // workspace layout
    unsigned short* Xhi = (unsigned short*)d_ws;
    unsigned short* Xlo = Xhi + nd;
    unsigned short* Yhi = Xlo + nd;
    unsigned short* Ylo = Yhi + nd;
    float* xx       = (float*)(Ylo + nd);
    float* yy       = xx + N;
    float* partials = yy + N;

    const int nBlk  = N / BLK;            // 64
    const int nPart = nBlk * nBlk;        // 4096

    lre_split_kernel<<<(2 * nd + 255) / 256, 256, 0, stream>>>(zx, zy, Xhi, Xlo, Yhi, Ylo, nd);
    lre_norms_kernel<<<(2 * N + 255) / 256, 256, 0, stream>>>(zx, zy, xx, yy, N);

    const size_t smem = (size_t)2 * BLK * ROWPITCH * sizeof(unsigned short); // ~68 KB of 320 KB LDS
    dim3 grid(nBlk, nBlk);
    lre_gemm_kernel<<<grid, 256, smem, stream>>>(Xhi, Xlo, Yhi, Ylo, xx, yy, partials, N);

    lre_finalize_kernel<<<1, 256, 0, stream>>>(partials, nPart, out, N);
}
